// SelfAttention_53437983097390
// MI455X (gfx1250) — matrix operand order repaired and checked by reading, not compiled, not run on hardware
//
#include <hip/hip_runtime.h>
#include <hip/hip_bf16.h>

// Problem dims (fixed by the reference)
#define B_ 8
#define C_ 256
#define N_ 4096
#define D_ 32

#define KP 40   // padded row pitch (shorts) for K tiles in LDS (80B, 16B-aligned)
#define VP 40   // padded row pitch (shorts) for V tiles in LDS

typedef __attribute__((ext_vector_type(16))) __bf16 v16bf;
typedef __attribute__((ext_vector_type(8)))  float  v8f;

union Frag { uint4 u[2]; v16bf v; };

__device__ __forceinline__ unsigned short f2bf(float f) {
  unsigned int u = __builtin_bit_cast(unsigned int, f);
  u += 0x7FFFu + ((u >> 16) & 1u);          // round-to-nearest-even
  return (unsigned short)(u >> 16);
}
__device__ __forceinline__ unsigned int pack2(float a, float b) {
  return (unsigned int)f2bf(a) | ((unsigned int)f2bf(b) << 16);
}

// Async DMA: global (16B) -> LDS, ASYNCcnt-tracked, saddr form:
// per-lane state is one constant 32-bit voffset; kb advance lives in SGPRs.
__device__ __forceinline__ void stage_async16(unsigned lds_dst, unsigned voff,
                                              unsigned long long sbase) {
  asm volatile("global_load_async_to_lds_b128 %0, %1, %2"
               :
               : "v"(lds_dst), "v"(voff), "s"(sbase)
               : "memory");
}

// ---------------------------------------------------------------------------
// Kernel 1: K = Wq*x, Q = Wk*x  ->  bf16, layout [B][N][32] (row per position)
// ---------------------------------------------------------------------------
__global__ __launch_bounds__(256) void proj_qk(
    const float* __restrict__ x, const float* __restrict__ Wq,
    const float* __restrict__ Wk, unsigned int* __restrict__ Km,
    unsigned int* __restrict__ Qm) {
  const int b = blockIdx.x >> 4;                     // N_/256 = 16 blocks/batch
  const int n = ((blockIdx.x & 15) << 8) + threadIdx.x;
  const float* xb = x + ((size_t)b * C_) * N_ + n;
  float ak[D_], aq[D_];
#pragma unroll
  for (int d = 0; d < D_; ++d) { ak[d] = 0.f; aq[d] = 0.f; }
  for (int c = 0; c < C_; ++c) {
    const float xv = xb[(size_t)c * N_];
#pragma unroll
    for (int d = 0; d < D_; ++d) {
      ak[d] = __builtin_fmaf(Wq[d * C_ + c], xv, ak[d]);  // keys  = f = Wq*x
      aq[d] = __builtin_fmaf(Wk[d * C_ + c], xv, aq[d]);  // query = g = Wk*x
    }
  }
  unsigned int* kr = Km + ((size_t)b * N_ + n) * (D_ / 2);
  unsigned int* qr = Qm + ((size_t)b * N_ + n) * (D_ / 2);
#pragma unroll
  for (int d = 0; d < D_ / 2; ++d) {
    kr[d] = pack2(ak[2 * d], ak[2 * d + 1]);
    qr[d] = pack2(aq[2 * d], aq[2 * d + 1]);
  }
}

// ---------------------------------------------------------------------------
// Kernel 2: h = Wv*x -> bf16, natural layout [B][C][N].
// ---------------------------------------------------------------------------
__global__ __launch_bounds__(256) void proj_v(
    const float* __restrict__ x, const float* __restrict__ Wv,
    unsigned short* __restrict__ hb) {
  const int cblk = blockIdx.x & 15;                  // C_/16
  const int nblk = (blockIdx.x >> 4) & 63;           // N_/64
  const int b    = blockIdx.x >> 10;
  const int c0   = cblk * 16 + (threadIdx.x >> 6) * 4;
  const int n    = nblk * 64 + (threadIdx.x & 63);
  const float* xb = x + ((size_t)b * C_) * N_ + n;
  float a0 = 0.f, a1 = 0.f, a2 = 0.f, a3 = 0.f;
  for (int c = 0; c < C_; ++c) {
    const float xv = xb[(size_t)c * N_];
    a0 = __builtin_fmaf(Wv[(c0 + 0) * C_ + c], xv, a0);
    a1 = __builtin_fmaf(Wv[(c0 + 1) * C_ + c], xv, a1);
    a2 = __builtin_fmaf(Wv[(c0 + 2) * C_ + c], xv, a2);
    a3 = __builtin_fmaf(Wv[(c0 + 3) * C_ + c], xv, a3);
  }
  hb[((size_t)b * C_ + c0 + 0) * N_ + n] = f2bf(a0);
  hb[((size_t)b * C_ + c0 + 1) * N_ + n] = f2bf(a1);
  hb[((size_t)b * C_ + c0 + 2) * N_ + n] = f2bf(a2);
  hb[((size_t)b * C_ + c0 + 3) * N_ + n] = f2bf(a3);
}

// ---------------------------------------------------------------------------
// Kernel 3: flash attention with double-buffered async K/V staging in LDS.
// Wave = 16 queries x 128 channels; 8 waves/block (4 query tiles x 2 halves).
// ---------------------------------------------------------------------------
__global__ __launch_bounds__(256) void flash_attn(
    const unsigned int* __restrict__ Qm, const unsigned int* __restrict__ Km,
    const unsigned short* __restrict__ hb, const float* __restrict__ x,
    const float* __restrict__ gamma, float* __restrict__ out) {
  __shared__ __align__(16) unsigned short Ks[2][32][KP];    //  5 KB
  __shared__ __align__(16) unsigned short Vs[2][C_][VP];    // 40 KB
  __shared__ __align__(16) unsigned short Plds[8][16][32];  //  8 KB

  const int tid  = threadIdx.x;
  const int b    = blockIdx.x >> 6;        // N_/64 = 64 blocks per batch
  const int qblk = blockIdx.x & 63;
  const int wave = tid >> 5;
  const int lane = tid & 31;
  const int hi   = lane >> 4;              // half-wave selector
  const int l15  = lane & 15;
  const int m0   = qblk * 64 + (wave >> 1) * 16;
  const int c0w  = (wave & 1) * 128;

  // Staging assignment: per-lane constant voffsets, kb advance in SGPRs.
  const int vrow   = tid >> 2;             // channel row 0..63 (+i*64)
  const int vchunk = (tid & 3) * 8;        // key offset in shorts (16B chunks)
  unsigned voffv[4], ldsv[4];
#pragma unroll
  for (int i = 0; i < 4; ++i) {
    voffv[i] = (unsigned)(((vrow + 64 * i) * N_ + vchunk) * 2);
    ldsv[i]  = (unsigned)(unsigned long long)&Vs[0][vrow + 64 * i][vchunk];
  }
  const unsigned voffk = (unsigned)((tid >> 2) * 64 + (tid & 3) * 16);
  const unsigned ldsk =
      (unsigned)(unsigned long long)&Ks[0][tid >> 2][(tid & 3) * 8];
  const unsigned bufstep_v = (unsigned)sizeof(unsigned short) * C_ * VP;
  const unsigned bufstep_k = (unsigned)sizeof(unsigned short) * 32 * KP;
  unsigned long long sbase_v =
      (unsigned long long)(hb + ((size_t)b * C_) * N_);
  unsigned long long sbase_k =
      (unsigned long long)(Km + ((size_t)b * N_) * 16);

  // Q A-fragment (16x32 bf16): lane holds row (l15), K-half chosen by hi.
  Frag qf;
  {
    const unsigned int* qr = Qm + ((size_t)b * N_ + m0 + l15) * 16;
    const int ko = hi * 8;                 // 16 bf16 = 8 dwords
    qf.u[0] = *(const uint4*)(qr + ko);
    qf.u[1] = *(const uint4*)(qr + 4 + ko);
  }

  // All-ones 32x16 bf16 B-fragment for the row-sum WMMA.
  Frag ones;
  ones.u[0] = make_uint4(0x3F803F80u, 0x3F803F80u, 0x3F803F80u, 0x3F803F80u);
  ones.u[1] = ones.u[0];

  v8f acc[8];
#pragma unroll
  for (int t = 0; t < 8; ++t) acc[t] = (v8f)0.0f;
  v8f accL = (v8f)0.0f;                    // softmax denominator tile
  float mrow[8];
#pragma unroll
  for (int r = 0; r < 8; ++r) mrow[r] = -3.0e38f;

  const v8f zf = (v8f)0.0f;

  // Prologue: stage key block 0 into buffer 0.
#pragma unroll
  for (int i = 0; i < 4; ++i) stage_async16(ldsv[i], voffv[i], sbase_v);
  if (tid < 128) stage_async16(ldsk, voffk, sbase_k);
  asm volatile("s_wait_asynccnt 0" ::: "memory");
  __syncthreads();

  int p = 0;
  for (int kb = 0; kb < N_; kb += 32) {
    // Kick off DMA of the next key block into the other buffer.
    if (kb + 32 < N_) {
      const unsigned bo_v = (p ^ 1) ? bufstep_v : 0u;
      const unsigned bo_k = (p ^ 1) ? bufstep_k : 0u;
#pragma unroll
      for (int i = 0; i < 4; ++i)
        stage_async16(ldsv[i] + bo_v, voffv[i], sbase_v + 64ull);
      if (tid < 128) stage_async16(ldsk + bo_k, voffk, sbase_k + 2048ull);
    }

    // Key B-fragments (32x16) straight from LDS.
    Frag kf0, kf1;
    {
      const unsigned short* k0 = &Ks[p][l15][hi * 16];
      kf0.u[0] = *(const uint4*)(k0);
      kf0.u[1] = *(const uint4*)(k0 + 8);
      const unsigned short* k1 = &Ks[p][16 + l15][hi * 16];
      kf1.u[0] = *(const uint4*)(k1);
      kf1.u[1] = *(const uint4*)(k1 + 8);
    }
    v8f s0 = __builtin_amdgcn_wmma_f32_16x16x32_bf16(
        false, qf.v, false, kf0.v, (short)0, zf, false, false);
    v8f s1 = __builtin_amdgcn_wmma_f32_16x16x32_bf16(
        false, qf.v, false, kf1.v, (short)0, zf, false, false);

    // Row max, level-major so all 8 swizzles per level are in flight at once.
    float vmax[8];
#pragma unroll
    for (int r = 0; r < 8; ++r) vmax[r] = fmaxf(s0[r], s1[r]);
#define SWZ_LEVEL(MASK)                                                       \
  {                                                                           \
    int tt[8];                                                                \
    _Pragma("unroll") for (int r = 0; r < 8; ++r) tt[r] =                     \
        __builtin_amdgcn_ds_swizzle(__builtin_bit_cast(int, vmax[r]), MASK);  \
    _Pragma("unroll") for (int r = 0; r < 8; ++r) vmax[r] =                   \
        fmaxf(vmax[r], __builtin_bit_cast(float, tt[r]));                     \
  }
    SWZ_LEVEL(0x041F)   // xor 1
    SWZ_LEVEL(0x081F)   // xor 2
    SWZ_LEVEL(0x101F)   // xor 4
    SWZ_LEVEL(0x201F)   // xor 8
#undef SWZ_LEVEL

    float scale[8], mnew[8];
    int any_change = 0;
#pragma unroll
    for (int r = 0; r < 8; ++r) {
      mnew[r] = fmaxf(mrow[r], vmax[r]);
      any_change |= (mnew[r] > mrow[r]) ? 1 : 0;
      scale[r] = __expf(mrow[r] - mnew[r]);
      mrow[r]  = mnew[r];
    }
    float p0[8], p1[8];
#pragma unroll
    for (int r = 0; r < 8; ++r) {
      p0[r] = __expf(s0[r] - mnew[r]);
      p1[r] = __expf(s1[r] - mnew[r]);
    }
    // Rescale only when some row's max actually grew (scale==1 otherwise).
    if (__any(any_change)) {
#pragma unroll
      for (int r = 0; r < 8; ++r) accL[r] *= scale[r];
#pragma unroll
      for (int t = 0; t < 8; ++t)
#pragma unroll
        for (int r = 0; r < 8; ++r) acc[t][r] *= scale[r];
    }

    // P: C/D layout -> LDS (bf16) -> reload in A layout (wave-private slab).
#pragma unroll
    for (int r = 0; r < 8; ++r) {
      const int row = r + hi * 8;
      Plds[wave][row][l15]      = f2bf(p0[r]);
      Plds[wave][row][16 + l15] = f2bf(p1[r]);
    }
    asm volatile("s_wait_dscnt 0" ::: "memory");
    Frag pf;
    {
      const unsigned short* pr = &Plds[wave][l15][hi * 16];
      pf.u[0] = *(const uint4*)(pr);
      pf.u[1] = *(const uint4*)(pr + 8);
    }

    // Denominator: accL += P * ones (row sums replicated in every column).
    accL = __builtin_amdgcn_wmma_f32_16x16x32_bf16(
        false, pf.v, false, ones.v, (short)0, accL, false, false);

    // O += P * V over this wave's 128 channels (8 tiles of 16), V from LDS.
#pragma unroll
    for (int t = 0; t < 8; ++t) {
      Frag vf;
      const unsigned short* vp = &Vs[p][c0w + t * 16 + l15][hi * 16];
      vf.u[0] = *(const uint4*)(vp);
      vf.u[1] = *(const uint4*)(vp + 8);
      acc[t] = __builtin_amdgcn_wmma_f32_16x16x32_bf16(
          false, pf.v, false, vf.v, (short)0, acc[t], false, false);
    }

    // Pipeline sync: own DMAs landed, all waves done with buffer p.
    asm volatile("s_wait_asynccnt 0" ::: "memory");
    __syncthreads();
    p ^= 1;
    sbase_v += 64ull;     // 32 keys * 2B along N
    sbase_k += 2048ull;   // 32 key rows * 64B
  }

  // Epilogue: out = acc * (gamma/l) + x  (v_rcp_f32 instead of IEEE divides).
  const float g = gamma[0];
  float rg[8];
#pragma unroll
  for (int r = 0; r < 8; ++r) rg[r] = g * __builtin_amdgcn_rcpf(accL[r]);
#pragma unroll
  for (int t = 0; t < 8; ++t) {
    const int c = c0w + t * 16 + l15;
#pragma unroll
    for (int r = 0; r < 8; ++r) {
      const int m = m0 + r + hi * 8;
      const size_t idx = ((size_t)b * C_ + c) * N_ + m;
      out[idx] = __builtin_fmaf(acc[t][r], rg[r], x[idx]);
    }
  }
}

// ---------------------------------------------------------------------------
extern "C" void kernel_launch(void* const* d_in, const int* in_sizes, int n_in,
                              void* d_out, int out_size, void* d_ws,
                              size_t ws_size, hipStream_t stream) {
  const float* x     = (const float*)d_in[0];
  const float* Wq    = (const float*)d_in[1];
  const float* Wk    = (const float*)d_in[2];
  const float* Wv    = (const float*)d_in[3];
  const float* gamma = (const float*)d_in[4];
  float* out = (float*)d_out;

  char* ws = (char*)d_ws;
  unsigned int*   Km = (unsigned int*)(ws);                        // 2 MB
  unsigned int*   Qm = (unsigned int*)(ws + ((size_t)2 << 20));    // 2 MB
  unsigned short* hb = (unsigned short*)(ws + ((size_t)4 << 20));  // 16 MB

  proj_qk<<<dim3(B_ * (N_ / 256)), dim3(256), 0, stream>>>(x, Wq, Wk, Km, Qm);
  proj_v<<<dim3(B_ * (N_ / 64) * (C_ / 16)), dim3(256), 0, stream>>>(x, Wv, hb);
  flash_attn<<<dim3(B_ * (N_ / 64)), dim3(256), 0, stream>>>(Qm, Km, hb, x,
                                                             gamma, out);
}
